// DijkstraGNN_14431090114819
// MI455X (gfx1250) — matrix operand order repaired and checked by reading, not compiled, not run on hardware
//
#include <hip/hip_runtime.h>

typedef __attribute__((ext_vector_type(16))) _Float16 v16h;
typedef __attribute__((ext_vector_type(8)))  float    v8f;

#define HD 64

// ---------- helpers ----------

__device__ __forceinline__ v8f wmma16(v16h a, v16h b, v8f c) {
  // D = A(16x32 f16) * B(32x16 f16) + C(16x16 f32)
  return __builtin_amdgcn_wmma_f32_16x16x32_f16(false, a, false, b, (short)0, c,
                                                false, false);
}

// wave-local LDS write->read ordering (CDNA5 split dep counters)
__device__ __forceinline__ void wave_lds_fence() {
  __builtin_amdgcn_wave_barrier();
  asm volatile("s_wait_dscnt 0" ::: "memory");
  __builtin_amdgcn_wave_barrier();
}

// monotonic float<->u32 mapping so GLOBAL_ATOMIC_MIN_U32 == float min
__device__ __forceinline__ unsigned fenc(float v) {
  unsigned b = __float_as_uint(v);
  return (b & 0x80000000u) ? ~b : (b | 0x80000000u);
}
__device__ __forceinline__ float fdec(unsigned k) {
  unsigned b = (k & 0x80000000u) ? (k & 0x7FFFFFFFu) : ~k;
  return __uint_as_float(b);
}

// Pack WMMA f16 A operand from an LDS-staged row-major tile.
// A 16x32 layout (contiguous, matches B packing): lane = row m, halves j:
//   g=0: K=j
//   g=1: K=16+j
__device__ __forceinline__ v16h pack_A(const _Float16* stg, int m, int g,
                                       int kbase, int ldk) {
  v16h a;
#pragma unroll
  for (int j = 0; j < 16; ++j) {
    int k = kbase + (g << 4) + j;
    a[j] = stg[m * ldk + k];
  }
  return a;
}

// ---------- tiny kernels ----------

__global__ void k_embed(const float* __restrict__ x, const float* __restrict__ w,
                        const float* __restrict__ b, _Float16* __restrict__ h,
                        int N) {
  int i = blockIdx.x * blockDim.x + threadIdx.x;
  if (i < N * HD) {
    int n = i >> 6, j = i & 63;
    h[i] = (_Float16)(x[n] * w[j] + b[j]);
  }
}

__global__ void k_init(unsigned* __restrict__ a, int n) {
  int i = blockIdx.x * blockDim.x + threadIdx.x;
  if (i < n) a[i] = 0xFF800000u;  // enc(+inf)
}

__global__ void k_out(const _Float16* __restrict__ h, const float* __restrict__ fw,
                      const float* __restrict__ fb, float* __restrict__ out, int N) {
  int i = blockIdx.x * blockDim.x + threadIdx.x;
  if (i < N) {
    float s = 0.f;
#pragma unroll
    for (int j = 0; j < HD; ++j) s += (float)h[(size_t)i * HD + j] * fw[j];
    out[i] = s + fb[0];
  }
}

// ---------- edge pass: m = relu([h_src|ea] W1 + b1) W2 + b2 ; atomic-min scatter ----------

__global__ __launch_bounds__(256) void k_edge(
    const _Float16* __restrict__ h, const int* __restrict__ src,
    const int* __restrict__ dst, const float* __restrict__ ea,
    const float* __restrict__ W1,  // (HD+1) x HD, row-major
    const float* __restrict__ b1, const float* __restrict__ W2,  // HD x HD
    const float* __restrict__ b2, unsigned* __restrict__ aggr, int E, int ntiles) {
  __shared__ _Float16 B1p[2 * 4 * 32 * 16];  // W1[0:64,:] packed in WMMA-B layout
  __shared__ _Float16 B2p[2 * 4 * 32 * 16];
  __shared__ float W1e[HD], b1s[HD], b2s[HD];
  __shared__ _Float16 stage[8][16 * HD];
  __shared__ float eaS[8][16];
  __shared__ int dstS[8][16];

  const int tid = threadIdx.x;
  // Dense-B layout: lane holds col n=lane%16; halves j -> K = j + 16*(lane/16)
  for (int p = tid; p < 2 * 4 * 32 * 16; p += blockDim.x) {
    int j = p & 15, lane = (p >> 4) & 31, nt = (p >> 9) & 3, s = p >> 11;
    int k = s * 32 + ((lane >> 4) << 4) + j;
    int n = nt * 16 + (lane & 15);
    B1p[p] = (_Float16)W1[k * HD + n];
    B2p[p] = (_Float16)W2[k * HD + n];
  }
  for (int p = tid; p < HD; p += blockDim.x) {
    W1e[p] = W1[64 * HD + p];  // edge_attr row of W1
    b1s[p] = b1[p];
    b2s[p] = b2[p];
  }
  __syncthreads();

  const int wave = tid >> 5, lane = tid & 31;
  const int g = lane >> 4, ml = lane & 15;
  _Float16* stg = stage[wave];
  const int gw = blockIdx.x * (blockDim.x >> 5) + wave;
  const int nw = gridDim.x * (blockDim.x >> 5);

  for (int tile = gw; tile < ntiles; tile += nw) {
    const int e0 = tile * 16;
    // prefetch next tile's edge metadata into cache (global_prefetch_b8)
    {
      int ne = e0 + nw * 16;
      if (ne < E) {
        __builtin_prefetch(&src[ne], 0, 0);
        __builtin_prefetch(&dst[ne], 0, 0);
        __builtin_prefetch(&ea[ne], 0, 0);
      }
    }
    {  // gather 16 f16 rows of h[src], 2 lanes per row, pure b128 bit-copies
      int m = lane >> 1, hf = lane & 1;
      int e = e0 + m;
      int si = (e < E) ? src[e] : 0;
      const uint4* hr = (const uint4*)(h + (size_t)si * HD + hf * 32);
      uint4* so = (uint4*)(stg + m * HD + hf * 32);
#pragma unroll
      for (int t = 0; t < 2; ++t) so[t] = hr[t];
      // stash per-tile edge_attr + dst once (kills guarded scalar loads in epilogues)
      if (lane < 16) {
        int e2 = e0 + lane;
        bool ok = e2 < E;
        eaS[wave][lane] = ok ? ea[e2] : 0.f;
        dstS[wave][lane] = ok ? dst[e2] : -1;
      }
    }
    wave_lds_fence();

    v8f zero = {0.f, 0.f, 0.f, 0.f, 0.f, 0.f, 0.f, 0.f};
    v8f acc[4] = {zero, zero, zero, zero};
#pragma unroll
    for (int s = 0; s < 2; ++s) {
      v16h a = pack_A(stg, ml, g, s * 32, HD);
      const v16h* Bp = (const v16h*)B1p;
#pragma unroll
      for (int nt = 0; nt < 4; ++nt)
        acc[nt] = wmma16(a, Bp[(s * 4 + nt) * 32 + lane], acc[nt]);
    }
    // per-lane copies of this tile's ea/dst (rows r+8g)
    float earr[8];
    int darr[8];
#pragma unroll
    for (int r = 0; r < 8; ++r) {
      earr[r] = eaS[wave][r + 8 * g];
      darr[r] = dstS[wave][r + 8 * g];
    }
    // epilogue 1: + b1 + ea * W1e, ReLU, restage as f16 (in-order LDS: reads
    // from GEMM1 are processed before these writes, no wait needed)
#pragma unroll
    for (int nt = 0; nt < 4; ++nt) {
      int n = nt * 16 + ml;
      float bb = b1s[n], we = W1e[n];
#pragma unroll
      for (int r = 0; r < 8; ++r) {
        int m = r + 8 * g;
        float v = acc[nt][r] + bb + earr[r] * we;
        stg[m * HD + n] = (_Float16)(v > 0.f ? v : 0.f);
      }
    }
    wave_lds_fence();

    v8f d[4] = {zero, zero, zero, zero};
#pragma unroll
    for (int s = 0; s < 2; ++s) {
      v16h a = pack_A(stg, ml, g, s * 32, HD);
      const v16h* Bp = (const v16h*)B2p;
#pragma unroll
      for (int nt = 0; nt < 4; ++nt)
        d[nt] = wmma16(a, Bp[(s * 4 + nt) * 32 + lane], d[nt]);
    }
    // epilogue 2: + b2, encoded atomic-min scatter to aggr[dst]
#pragma unroll
    for (int nt = 0; nt < 4; ++nt) {
      int n = nt * 16 + ml;
      float bb = b2s[n];
#pragma unroll
      for (int r = 0; r < 8; ++r) {
        int di = darr[r];
        if (di >= 0)
          atomicMin(&aggr[(size_t)di * HD + n], fenc(d[nt][r] + bb));
      }
    }
  }
}

// ---------- node pass: h' = relu([h|aggr] U1 + b1) U2 + b2 ----------

__global__ __launch_bounds__(256) void k_update(
    const _Float16* __restrict__ h, const unsigned* __restrict__ aggr,
    const float* __restrict__ U1,  // 2*HD x HD
    const float* __restrict__ ub1, const float* __restrict__ U2,  // HD x HD
    const float* __restrict__ ub2, _Float16* __restrict__ hout, int N, int ntiles) {
  __shared__ _Float16 B1p[4 * 4 * 32 * 16];
  __shared__ _Float16 B2p[2 * 4 * 32 * 16];
  __shared__ float b1s[HD], b2s[HD];
  __shared__ _Float16 stage[8][16 * 128];

  const int tid = threadIdx.x;
  for (int p = tid; p < 4 * 4 * 32 * 16; p += blockDim.x) {
    int j = p & 15, lane = (p >> 4) & 31, nt = (p >> 9) & 3, s = p >> 11;
    int k = s * 32 + ((lane >> 4) << 4) + j;
    int n = nt * 16 + (lane & 15);
    B1p[p] = (_Float16)U1[k * HD + n];
  }
  for (int p = tid; p < 2 * 4 * 32 * 16; p += blockDim.x) {
    int j = p & 15, lane = (p >> 4) & 31, nt = (p >> 9) & 3, s = p >> 11;
    int k = s * 32 + ((lane >> 4) << 4) + j;
    int n = nt * 16 + (lane & 15);
    B2p[p] = (_Float16)U2[k * HD + n];
  }
  for (int p = tid; p < HD; p += blockDim.x) {
    b1s[p] = ub1[p];
    b2s[p] = ub2[p];
  }
  __syncthreads();

  const int wave = tid >> 5, lane = tid & 31;
  const int g = lane >> 4, ml = lane & 15;
  _Float16* stg = stage[wave];
  const int gw = blockIdx.x * (blockDim.x >> 5) + wave;
  const int nw = gridDim.x * (blockDim.x >> 5);

  for (int tile = gw; tile < ntiles; tile += nw) {
    const int n0 = tile * 16;
    {  // stage [h | decoded aggr] as 16x128 f16
      int m = lane >> 1, hf = lane & 1;
      int node = n0 + m;
      if (node >= N) node = N - 1;
      const uint4* hr = (const uint4*)(h + (size_t)node * HD + hf * 32);
      const uint4* ar = (const uint4*)(aggr + (size_t)node * HD + hf * 32);
      uint4* so16 = (uint4*)(stg + m * 128 + hf * 32);
      _Float16* so = stg + m * 128 + hf * 32;
#pragma unroll
      for (int t = 0; t < 2; ++t) so16[t] = hr[t];  // h is already f16
#pragma unroll
      for (int t = 0; t < 8; ++t) {
        uint4 u = ar[t];
        float f0 = fdec(u.x), f1 = fdec(u.y), f2 = fdec(u.z), f3 = fdec(u.w);
        // empty segments / non-finite -> 0 (matches reference isfinite mask)
        if (!(f0 > -3.0e38f && f0 < 3.0e38f)) f0 = 0.f;
        if (!(f1 > -3.0e38f && f1 < 3.0e38f)) f1 = 0.f;
        if (!(f2 > -3.0e38f && f2 < 3.0e38f)) f2 = 0.f;
        if (!(f3 > -3.0e38f && f3 < 3.0e38f)) f3 = 0.f;
        so[64 + t * 4 + 0] = (_Float16)f0;
        so[64 + t * 4 + 1] = (_Float16)f1;
        so[64 + t * 4 + 2] = (_Float16)f2;
        so[64 + t * 4 + 3] = (_Float16)f3;
      }
    }
    wave_lds_fence();

    v8f zero = {0.f, 0.f, 0.f, 0.f, 0.f, 0.f, 0.f, 0.f};
    v8f acc[4] = {zero, zero, zero, zero};
#pragma unroll
    for (int s = 0; s < 4; ++s) {
      v16h a = pack_A(stg, ml, g, s * 32, 128);
      const v16h* Bp = (const v16h*)B1p;
#pragma unroll
      for (int nt = 0; nt < 4; ++nt)
        acc[nt] = wmma16(a, Bp[(s * 4 + nt) * 32 + lane], acc[nt]);
    }
    // epilogue 1: + b1, ReLU, restage (in-order LDS: no wait needed here)
#pragma unroll
    for (int nt = 0; nt < 4; ++nt) {
      int n = nt * 16 + ml;
      float bb = b1s[n];
#pragma unroll
      for (int r = 0; r < 8; ++r) {
        int m = r + 8 * g;
        float v = acc[nt][r] + bb;
        stg[m * 128 + n] = (_Float16)(v > 0.f ? v : 0.f);
      }
    }
    wave_lds_fence();

    v8f d[4] = {zero, zero, zero, zero};
#pragma unroll
    for (int s = 0; s < 2; ++s) {
      v16h a = pack_A(stg, ml, g, s * 32, 128);
      const v16h* Bp = (const v16h*)B2p;
#pragma unroll
      for (int nt = 0; nt < 4; ++nt)
        d[nt] = wmma16(a, Bp[(s * 4 + nt) * 32 + lane], d[nt]);
    }
    const bool full = (n0 + 16 <= N);
#pragma unroll
    for (int nt = 0; nt < 4; ++nt) {
      int n = nt * 16 + ml;
      float bb = b2s[n];
#pragma unroll
      for (int r = 0; r < 8; ++r) {
        int m = r + 8 * g;
        int node = n0 + m;
        if (full || node < N)
          hout[(size_t)node * HD + n] = (_Float16)(d[nt][r] + bb);
      }
    }
  }
}

// ---------- launch ----------

extern "C" void kernel_launch(void* const* d_in, const int* in_sizes, int n_in,
                              void* d_out, int out_size, void* d_ws, size_t ws_size,
                              hipStream_t stream) {
  (void)n_in;
  (void)out_size;
  (void)ws_size;
  const float* x = (const float*)d_in[0];
  const int* ei = (const int*)d_in[1];
  const float* ea = (const float*)d_in[2];
  const float* emb_w = (const float*)d_in[3];
  const float* emb_b = (const float*)d_in[4];
  const float* msg_w1 = (const float*)d_in[5];
  const float* msg_b1 = (const float*)d_in[6];
  const float* msg_w2 = (const float*)d_in[7];
  const float* msg_b2 = (const float*)d_in[8];
  const float* upd_w1 = (const float*)d_in[9];
  const float* upd_b1 = (const float*)d_in[10];
  const float* upd_w2 = (const float*)d_in[11];
  const float* upd_b2 = (const float*)d_in[12];
  const float* fc_w = (const float*)d_in[13];
  const float* fc_b = (const float*)d_in[14];

  const int N = in_sizes[0];       // NF == 1
  const int E = in_sizes[2];       // EF == 1
  const int L = in_sizes[6] / HD;  // msg_b1 is L*HD

  _Float16* hA = (_Float16*)d_ws;
  _Float16* hB = hA + (size_t)N * HD;
  unsigned* aggr = (unsigned*)(hB + (size_t)N * HD);

  const int etiles = (E + 15) / 16;
  const int vtiles = (N + 15) / 16;

  k_embed<<<(N * HD + 255) / 256, 256, 0, stream>>>(x, emb_w, emb_b, hA, N);

  const _Float16* hc = hA;
  _Float16* hn = hB;
  for (int l = 0; l < L; ++l) {
    k_init<<<(N * HD + 255) / 256, 256, 0, stream>>>(aggr, N * HD);
    k_edge<<<1024, 256, 0, stream>>>(
        hc, ei, ei + E, ea, msg_w1 + (size_t)l * (HD + 1) * HD,
        msg_b1 + (size_t)l * HD, msg_w2 + (size_t)l * HD * HD,
        msg_b2 + (size_t)l * HD, aggr, E, etiles);
    k_update<<<512, 256, 0, stream>>>(
        hc, aggr, upd_w1 + (size_t)l * 2 * HD * HD, upd_b1 + (size_t)l * HD,
        upd_w2 + (size_t)l * HD * HD, upd_b2 + (size_t)l * HD, hn, N, vtiles);
    const _Float16* t = hc;
    hc = hn;
    hn = (_Float16*)t;
  }
  k_out<<<(N + 255) / 256, 256, 0, stream>>>(hc, fc_w, fc_b, (float*)d_out, N);
}
